// Attention_28183575396711
// MI455X (gfx1250) — matrix operand order repaired and checked by reading, not compiled, not run on hardware
//
#include <hip/hip_runtime.h>
#include <hip/hip_bf16.h>
#include <stdint.h>

// Problem dims
static constexpr int BATCH = 64;
static constexpr int SEQ   = 2048;
static constexpr int RNND  = 1024;
static constexpr int EMBD  = 512;
static constexpr int ATTD  = 128;
static constexpr int NFD   = 32;
static constexpr int KW    = 31;
static constexpr int PADW  = 15;

typedef __bf16 bf16_t;
typedef bf16_t v16bf __attribute__((ext_vector_type(16)));
typedef float  v8f   __attribute__((ext_vector_type(8)));

__device__ __forceinline__ bf16_t f2bf(float f) {
  // round-to-nearest-even f32 -> bf16
  unsigned int u = __builtin_bit_cast(unsigned int, f);
  unsigned int r = (u + 0x7FFFu + ((u >> 16) & 1u)) >> 16;
  unsigned short s = (unsigned short)r;
  return __builtin_bit_cast(bf16_t, s);
}

// Branch-free tanh: exact at both saturation ends (exp->inf => 1, exp->0 => -1).
// Lowers to v_exp_f32 + v_rcp_f32 + 3 VALU, no EXEC divergence.
__device__ __forceinline__ float fast_tanh(float x) {
  float ex = __expf(2.0f * x);
  return 1.0f - 2.0f * __builtin_amdgcn_rcpf(ex + 1.0f);
}

// ---------------------------------------------------------------------------
// Kernel 1: pq[b,a] = h[b,:] . Wq[a,:] + bq[a]     (64 x 1024 x 128, tiny)
// ---------------------------------------------------------------------------
__global__ __launch_bounds__(128) void pq_kernel(
    const float* __restrict__ h, const float* __restrict__ Wq,
    const float* __restrict__ bq, float* __restrict__ pq) {
  __shared__ float sh[RNND];
  const int b = blockIdx.x, a = threadIdx.x;
  for (int i = threadIdx.x; i < RNND; i += 128) sh[i] = h[b * RNND + i];
  __syncthreads();
  const float4* wr = (const float4*)(Wq + (size_t)a * RNND);
  float acc = 0.f;
#pragma unroll 4
  for (int k4 = 0; k4 < RNND / 4; ++k4) {
    float4 w = wr[k4];
    acc += w.x * sh[4 * k4 + 0] + w.y * sh[4 * k4 + 1] +
           w.z * sh[4 * k4 + 2] + w.w * sh[4 * k4 + 3];
  }
  pq[b * ATTD + a] = acc + bq[a];
}

// ---------------------------------------------------------------------------
// Kernel 2: conv1d + loc projection (WMMA bf16 K=32) + tanh-energy -> energies
// One wave per (b, 16-t tile). A = conv features (16t x 32f), B = Wl^T (32x16).
// ---------------------------------------------------------------------------
__global__ __launch_bounds__(32) void energy_kernel(
    const float* __restrict__ attw, const float* __restrict__ pm,
    const unsigned char* __restrict__ mask,
    const float* __restrict__ conv_w, const float* __restrict__ conv_b,
    const float* __restrict__ Wl, const float* __restrict__ bl,
    const float* __restrict__ vw, const float* __restrict__ vb,
    const float* __restrict__ pq, float* __restrict__ energies) {
  __shared__ float s_w[NFD * 2 * KW];  // 1984 floats
  __shared__ float s_b[NFD];

  const int l  = threadIdx.x;       // 0..31
  const int t0 = blockIdx.x * 16;
  const int b  = blockIdx.y;

  for (int i = l; i < NFD * 2 * KW; i += 32) s_w[i] = conv_w[i];
  s_b[l] = conv_b[l];               // NFD == 32
  __syncthreads();

  const int half   = l >> 4;        // 0: lanes 0-15, 1: lanes 16-31
  const int lane16 = l & 15;
  const int tA     = t0 + lane16;   // A-matrix row this lane supplies

  // Stage attention-weight window [tA-15, tA+15] for both channels in VGPRs.
  float aw[2][KW];
#pragma unroll
  for (int c = 0; c < 2; ++c)
#pragma unroll
    for (int k = 0; k < KW; ++k) {
      int idx = tA + k - PADW;
      aw[c][k] = (idx >= 0 && idx < SEQ) ? attw[((size_t)b * 2 + c) * SEQ + idx] : 0.f;
    }

  // Build A fragment: ISA layout, lane<16 holds K={0..7,16..23}, lane>=16 K={8..15,24..31}
  v16bf afrag;
#pragma unroll
  for (int e = 0; e < 16; ++e) {
    int f = ((e < 8) ? e : e + 8) + half * 8;
    float acc = s_b[f];
    const float* w0 = &s_w[(f * 2 + 0) * KW];
    const float* w1 = &s_w[(f * 2 + 1) * KW];
#pragma unroll
    for (int k = 0; k < KW; ++k) acc += w0[k] * aw[0][k] + w1[k] * aw[1][k];
    afrag[e] = f2bf(acc);
  }

  float partial[8];
#pragma unroll
  for (int r = 0; r < 8; ++r) partial[r] = 0.f;
  const int tD = t0 + half * 8;     // D rows covered by this lane: tD + r

#pragma unroll
  for (int a0 = 0; a0 < ATTD; a0 += 16) {
    const int a = a0 + lane16;
    // B fragment: lane = output column a; element e -> K = ((e<8)?e:e+8) + 8*half
    v16bf bfrag;
    const float* wlrow = Wl + (size_t)a * NFD;
#pragma unroll
    for (int e = 0; e < 16; ++e) bfrag[e] = f2bf(wlrow[((e < 8) ? e : e + 8) + half * 8]);

    v8f c = {};
    v8f d = __builtin_amdgcn_wmma_f32_16x16x32_bf16(
        false, afrag, false, bfrag, (short)0, c, false, false);

    const float base = pq[b * ATTD + a] + bl[a];
    const float va   = vw[a];
#pragma unroll
    for (int r = 0; r < 8; ++r) {
      const int t = tD + r;
      float val = d[r] + base + pm[((size_t)b * SEQ + t) * ATTD + a];
      partial[r] += va * fast_tanh(val);
    }
  }

  // Reduce the 128 attention dims: sum across the 16 lanes of each half-wave.
#pragma unroll
  for (int m = 1; m <= 8; m <<= 1)
#pragma unroll
    for (int r = 0; r < 8; ++r) partial[r] += __shfl_xor(partial[r], m, 32);

  if (lane16 == 0) {
    const float vbv = vb[0];
#pragma unroll
    for (int r = 0; r < 8; ++r) {
      const int t = tD + r;
      float e = partial[r] + vbv;
      if (mask[(size_t)b * SEQ + t]) e = -INFINITY;
      energies[(size_t)b * SEQ + t] = e;
    }
  }
}

// ---------------------------------------------------------------------------
// Kernel 3: softmax over T per batch row -> attention_weights (d_out region)
// ---------------------------------------------------------------------------
__global__ __launch_bounds__(256) void softmax_kernel(
    const float* __restrict__ energies, float* __restrict__ weights) {
  __shared__ float red[256];
  const int b = blockIdx.x, tid = threadIdx.x;
  const float* e = energies + (size_t)b * SEQ;

  float mx = -INFINITY;
  for (int t = tid; t < SEQ; t += 256) mx = fmaxf(mx, e[t]);
  red[tid] = mx;
  __syncthreads();
  for (int s = 128; s > 0; s >>= 1) {
    if (tid < s) red[tid] = fmaxf(red[tid], red[tid + s]);
    __syncthreads();
  }
  mx = red[0];
  __syncthreads();

  float sum = 0.f;
  for (int t = tid; t < SEQ; t += 256) sum += __expf(e[t] - mx);
  red[tid] = sum;
  __syncthreads();
  for (int s = 128; s > 0; s >>= 1) {
    if (tid < s) red[tid] += red[tid + s];
    __syncthreads();
  }
  const float inv = 1.f / red[0];
  for (int t = tid; t < SEQ; t += 256)
    weights[(size_t)b * SEQ + t] = __expf(e[t] - mx) * inv;
}

// ---------------------------------------------------------------------------
// Kernel 4a/4b: context[b,e] = sum_t w[b,t] * memory[b,t,e]  (HBM-bound pass)
// ---------------------------------------------------------------------------
__global__ void zero_kernel(float* __restrict__ p, int n) {
  int i = blockIdx.x * blockDim.x + threadIdx.x;
  if (i < n) p[i] = 0.f;
}

__global__ __launch_bounds__(256) void context_kernel(
    const float* __restrict__ weights, const float* __restrict__ memory,
    float* __restrict__ ctx) {
  const int b   = blockIdx.x;
  const int t0  = blockIdx.y * 512;
  const int tid = threadIdx.x;
  float acc0 = 0.f, acc1 = 0.f;
  const float* wrow = weights + (size_t)b * SEQ + t0;
  const float* mrow = memory + ((size_t)b * SEQ + t0) * EMBD;
  for (int t = 0; t < 512; ++t) {
    const float wt = wrow[t];
    const float* m = mrow + (size_t)t * EMBD;
    __builtin_prefetch(m + 8 * EMBD, 0, 1);   // global_prefetch_b8 ahead of stream
    acc0 += wt * m[tid];
    acc1 += wt * m[tid + 256];
  }
  atomicAdd(&ctx[b * EMBD + tid], acc0);
  atomicAdd(&ctx[b * EMBD + tid + 256], acc1);
}

// ---------------------------------------------------------------------------
extern "C" void kernel_launch(void* const* d_in, const int* in_sizes, int n_in,
                              void* d_out, int out_size, void* d_ws, size_t ws_size,
                              hipStream_t stream) {
  const float* h    = (const float*)d_in[0];   // [B, RNN]
  const float* mem  = (const float*)d_in[1];   // [B, T, EMB]
  const float* pm   = (const float*)d_in[2];   // [B, T, ATT]
  const float* attw = (const float*)d_in[3];   // [B, 2, T]
  const unsigned char* mask = (const unsigned char*)d_in[4]; // [B, T] bool
  const float* Wq = (const float*)d_in[5];     // [ATT, RNN]
  const float* bq = (const float*)d_in[6];     // [ATT]
  const float* cw = (const float*)d_in[7];     // [NF, 2, K]
  const float* cb = (const float*)d_in[8];     // [NF]
  const float* Wl = (const float*)d_in[9];     // [ATT, NF]
  const float* bl = (const float*)d_in[10];    // [ATT]
  const float* vw = (const float*)d_in[11];    // [1, ATT]
  const float* vb = (const float*)d_in[12];    // [1]

  float* out_ctx = (float*)d_out;              // [B, EMB]  (output 0)
  float* out_w   = out_ctx + BATCH * EMBD;     // [B, T]    (output 1)

  float* pq       = (float*)d_ws;              // [B, ATT]
  float* energies = pq + BATCH * ATTD;         // [B, T]

  pq_kernel<<<BATCH, 128, 0, stream>>>(h, Wq, bq, pq);
  energy_kernel<<<dim3(SEQ / 16, BATCH), 32, 0, stream>>>(
      attw, pm, mask, cw, cb, Wl, bl, vw, vb, pq, energies);
  softmax_kernel<<<BATCH, 256, 0, stream>>>(energies, out_w);
  zero_kernel<<<(BATCH * EMBD + 255) / 256, 256, 0, stream>>>(out_ctx, BATCH * EMBD);
  context_kernel<<<dim3(BATCH, SEQ / 512), 256, 0, stream>>>(out_w, mem, out_ctx);
}
